// DiscreteFlow_42769284334037
// MI455X (gfx1250) — compile-verified, matrix operand order checked by reading, not run on hardware
//
#include <hip/hip_runtime.h>
#include <hip/hip_bf16.h>

// ---------------- problem constants ----------------
constexpr int Bb     = 32;
constexpr int Dd     = 256;     // sequence length
constexpr int Kk     = 128;     // vocab
constexpr int DIMc   = 512;
constexpr int HEADSc = 8;
constexpr int HDc    = 64;
constexpr int DEPTHc = 4;
constexpr int NFc    = 4;
constexpr int Mrows  = Bb * Dd; // 8192 tokens

typedef __bf16 bf16;
typedef bf16  bf16x8  __attribute__((ext_vector_type(8)));
typedef bf16  bf16x16 __attribute__((ext_vector_type(16)));
typedef float floatx8 __attribute__((ext_vector_type(8)));

__device__ __forceinline__ bf16 f2bf(float f) {
    unsigned u = __builtin_bit_cast(unsigned, f);
    unsigned r = (u + 0x7FFFu + ((u >> 16) & 1u)) >> 16;
    unsigned short hs = (unsigned short)r;
    return __builtin_bit_cast(bf16, hs);
}

// ---------------- generic WMMA GEMM ----------------
// C[m,n] = sum_k A[m,k] * W[n,k]
//   A: MxK f32 row-major (activations, converted f32->bf16 while staging to LDS)
//   W: NxK bf16 row-major (pre-converted weights, staged via async-to-LDS DMA)
// Epilogue (compile-time MODE):
//  0: C = acc
//  1: C = acc + aux[(m % D)*N + n]        (pos-embed)
//  2: C = acc + aux[n]                    (bias)
//  3: C += acc                            (residual accumulate)
//  4: C = relu(acc)^2                     (MLP activation, fused)
#define GBM 128
#define GBN 128
#define GBK 32
#define LSTR 40   // bf16 elements per LDS row (80B: 16B aligned, conflict-friendly)

template <int MODE>
__global__ __launch_bounds__(256)
void k_gemm(const float* __restrict__ A, const bf16* __restrict__ W,
            float* __restrict__ C, const float* __restrict__ aux,
            int M, int N, int K) {
    __shared__ bf16 As[GBM * LSTR];
    __shared__ bf16 Bs[GBN * LSTR];

    const int bm = blockIdx.x * GBM;
    const int bn = blockIdx.y * GBN;
    const int t  = threadIdx.x;
    const int w  = t >> 5;          // wave id 0..7
    const int l  = t & 31;
    const int l16 = l & 15;
    const int lhi = l >> 4;         // 0 or 1
    const int wm = w & 3;           // row group: wm*32
    const int wn = w >> 2;          // col group: wn*64

    const int sr = t >> 3;          // A staging row (0..31), +32 strides
    const int sc = (t & 7) * 4;     // A staging k offset

    // B staging via async DMA: 128 rows x 64B = 512 x 16B chunks, 2 per thread
    const int bchunk = t * 2;

    floatx8 acc[2][4];
#pragma unroll
    for (int mt = 0; mt < 2; ++mt)
#pragma unroll
        for (int nt = 0; nt < 4; ++nt)
            acc[mt][nt] = (floatx8)0.0f;

    for (int k0 = 0; k0 < K; k0 += GBK) {
        // ---- stage B tile: raw bf16 global -> LDS via async DMA ----
#pragma unroll
        for (int cc = 0; cc < 2; ++cc) {
            const int chunk = bchunk + cc;
            const int r = chunk >> 2;        // 0..127
            const int c = chunk & 3;         // 16B chunk within 64B row
            unsigned lds_off = (unsigned)(uintptr_t)(&Bs[r * LSTR]) + c * 16;
            const char* gsrc = (const char*)(W + (size_t)(bn + r) * K + k0) + c * 16;
            unsigned long long ga = (unsigned long long)(uintptr_t)gsrc;
            asm volatile("global_load_async_to_lds_b128 %0, %1, off"
                         :: "v"(lds_off), "v"(ga) : "memory");
        }
        // ---- stage A tile: f32 global -> bf16 LDS (VALU convert) ----
#pragma unroll
        for (int rr = 0; rr < 4; ++rr) {
            const int r = sr + rr * 32;
            const float4 av = *(const float4*)(A + (size_t)(bm + r) * K + k0 + sc);
            bf16* da = &As[r * LSTR + sc];
            da[0] = f2bf(av.x); da[1] = f2bf(av.y); da[2] = f2bf(av.z); da[3] = f2bf(av.w);
        }
        asm volatile("s_wait_asynccnt 0x0" ::: "memory");
        __syncthreads();

        // ---- build fragments (ISA 16-bit A/B lane layout) ----
        bf16x16 af[2];
#pragma unroll
        for (int mt = 0; mt < 2; ++mt) {
            const bf16* base = &As[(wm * 32 + mt * 16 + l16) * LSTR + lhi * 8];
            bf16x8 lo = *(const bf16x8*)(base);
            bf16x8 hi = *(const bf16x8*)(base + 16);
#pragma unroll
            for (int i = 0; i < 8; ++i) { af[mt][i] = lo[i]; af[mt][i + 8] = hi[i]; }
        }
        bf16x16 bfr[4];
#pragma unroll
        for (int nt = 0; nt < 4; ++nt) {
            const bf16* base = &Bs[(wn * 64 + nt * 16 + l16) * LSTR + lhi * 8];
            bf16x8 lo = *(const bf16x8*)(base);
            bf16x8 hi = *(const bf16x8*)(base + 16);
#pragma unroll
            for (int i = 0; i < 8; ++i) { bfr[nt][i] = lo[i]; bfr[nt][i + 8] = hi[i]; }
        }

        // ---- WMMA ----
#pragma unroll
        for (int mt = 0; mt < 2; ++mt)
#pragma unroll
            for (int nt = 0; nt < 4; ++nt)
                acc[mt][nt] = __builtin_amdgcn_wmma_f32_16x16x32_bf16(
                    false, af[mt], false, bfr[nt], (short)0, acc[mt][nt], false, false);
        __syncthreads();
    }

    // ---- straight-line epilogue (no runtime branching) ----
#pragma unroll
    for (int mt = 0; mt < 2; ++mt) {
#pragma unroll
        for (int nt = 0; nt < 4; ++nt) {
            const int n = bn + wn * 64 + nt * 16 + l16;
            const int mbase = bm + wm * 32 + mt * 16 + lhi * 8;
            float* cp = C + (size_t)mbase * N + n;
            const float* auxp = nullptr;
            if constexpr (MODE == 1) auxp = aux + (size_t)(mbase & (Dd - 1)) * N + n;
#pragma unroll
            for (int r = 0; r < 8; ++r) {
                const float v = acc[mt][nt][r];
                if constexpr (MODE == 0) {
                    cp[(size_t)r * N] = v;
                } else if constexpr (MODE == 1) {
                    cp[(size_t)r * N] = v + auxp[(size_t)r * N];
                } else if constexpr (MODE == 2) {
                    cp[(size_t)r * N] = v + aux[n];
                } else if constexpr (MODE == 3) {
                    cp[(size_t)r * N] += v;
                } else { // 4: relu^2
                    const float f = fmaxf(v, 0.0f);
                    cp[(size_t)r * N] = f * f;
                }
            }
        }
    }
}

// ---------------- weight f32 -> bf16 pre-conversion ----------------
__global__ __launch_bounds__(256)
void k_cvt_bf16(const float* __restrict__ in, bf16* __restrict__ out) {
    const size_t i = ((size_t)blockIdx.x * 256 + threadIdx.x) * 4;
    const float4 v = *(const float4*)(in + i);
    bf16* o = out + i;
    o[0] = f2bf(v.x); o[1] = f2bf(v.y); o[2] = f2bf(v.z); o[3] = f2bf(v.w);
}

// ---------------- elementwise / small kernels ----------------
__global__ void k_onehot(const int* __restrict__ y, float* __restrict__ X) {
    const int m = blockIdx.x, k = threadIdx.x;
    X[(size_t)m * Kk + k] = (y[m] == k) ? 1.0f : 0.0f;
}

__global__ void k_mask(const float* __restrict__ X, float* __restrict__ xm, int parity) {
    const int m = blockIdx.x, k = threadIdx.x;
    const int d = m & (Dd - 1);
    const float mv = ((d & 1) == parity) ? 1.0f : 0.0f;
    xm[(size_t)m * Kk + k] = mv * X[(size_t)m * Kk + k];
}

__global__ __launch_bounds__(256)
void k_rms(const float* __restrict__ in, float* __restrict__ out, int width) {
    __shared__ float red[8];
    const int row = blockIdx.x;
    const float* p = in + (size_t)row * width;
    float s = 0.0f;
    for (int j = threadIdx.x; j < width; j += 256) { float v = p[j]; s += v * v; }
    for (int off = 16; off; off >>= 1) s += __shfl_xor(s, off, 32);
    if ((threadIdx.x & 31) == 0) red[threadIdx.x >> 5] = s;
    __syncthreads();
    if (threadIdx.x == 0) {
        float tot = 0.0f;
        for (int i = 0; i < 8; ++i) tot += red[i];
        red[0] = rsqrtf(tot / (float)width + 1e-6f);
    }
    __syncthreads();
    const float r = red[0];
    for (int j = threadIdx.x; j < width; j += 256) out[(size_t)row * width + j] = p[j] * r;
}

// per-head RMSNorm + RoPE on q and k, in-place on the qkv buffer (M x 1536)
__global__ void k_rope(float* __restrict__ qkv) {
    const int idx = blockIdx.x;          // ((m*8 + h)*2 + qk)
    const int qk = idx & 1;
    const int mh = idx >> 1;
    const int h = mh & 7;
    const int m = mh >> 3;
    const int pos = m & (Dd - 1);
    float* row = qkv + (size_t)m * 1536 + qk * 512 + h * 64;
    const int j = threadIdx.x;           // pair index 0..31
    float x0 = row[2 * j], x1 = row[2 * j + 1];
    float ss = x0 * x0 + x1 * x1;
    for (int off = 16; off; off >>= 1) ss += __shfl_xor(ss, off, 32);
    const float r = rsqrtf(ss / 64.0f + 1e-6f);
    x0 *= r; x1 *= r;
    const float freq = __powf(10000.0f, -(float)j / 32.0f);
    float sn, cs;
    __sincosf((float)pos * freq, &sn, &cs);
    row[2 * j]     = x0 * cs - x1 * sn;
    row[2 * j + 1] = x0 * sn + x1 * cs;
}

// one workgroup per (b,h): online-softmax attention, K tile in LDS
__global__ __launch_bounds__(256)
void k_attn(const float* __restrict__ qkv, float* __restrict__ o) {
    extern __shared__ float Ks[];        // 256 x 64 f32 = 64KB
    const int bh = blockIdx.x;
    const int h = bh & 7;
    const int b = bh >> 3;
    const float* base = qkv + (size_t)b * Dd * 1536;
    for (int e = threadIdx.x; e < Dd * HDc; e += 256) {
        const int srow = e >> 6, j = e & 63;
        Ks[e] = base[(size_t)srow * 1536 + 512 + h * 64 + j];
    }
    __syncthreads();
    const int sq = threadIdx.x;          // one q row per thread
    float q[64];
    const float* qp = base + (size_t)sq * 1536 + h * 64;
#pragma unroll
    for (int j = 0; j < 64; ++j) q[j] = qp[j];
    float mmax = -1e30f, lsum = 0.0f, oacc[64];
#pragma unroll
    for (int j = 0; j < 64; ++j) oacc[j] = 0.0f;
    for (int kk = 0; kk < Dd; ++kk) {
        const float* kr = &Ks[kk * 64];
        float s = 0.0f;
#pragma unroll
        for (int j = 0; j < 64; ++j) s += q[j] * kr[j];
        s *= 0.125f;                      // 1/sqrt(64)
        const float nm = fmaxf(mmax, s);
        const float corr = __expf(mmax - nm);
        const float p = __expf(s - nm);
        lsum = lsum * corr + p;
        const float* vr = base + (size_t)kk * 1536 + 1024 + h * 64;
#pragma unroll
        for (int j = 0; j < 64; ++j) oacc[j] = oacc[j] * corr + p * vr[j];
        mmax = nm;
    }
    const float inv = 1.0f / lsum;
    float* op = o + ((size_t)(b * Dd + sq) * DIMc) + h * 64;
#pragma unroll
    for (int j = 0; j < 64; ++j) op[j] = oacc[j] * inv;
}

__global__ void k_skip_add(float* __restrict__ H, const float* __restrict__ S,
                           const float* __restrict__ alpha) {
    const size_t i = (size_t)blockIdx.x * 256 + threadIdx.x;
    const float a = alpha[0];
    const float sig = 1.0f / (1.0f + __expf(-a));
    H[i] += sig * S[i];
}

__global__ void k_argmax(const float* __restrict__ logits, int* __restrict__ am) {
    __shared__ float bv[128];
    __shared__ int   bi[128];
    const int m = blockIdx.x, k = threadIdx.x;
    bv[k] = logits[(size_t)m * Kk + k];
    bi[k] = k;
    __syncthreads();
    for (int off = 64; off; off >>= 1) {
        if (k < off && bv[k + off] > bv[k]) { bv[k] = bv[k + off]; bi[k] = bi[k + off]; }
        __syncthreads();
    }
    if (k == 0) am[m] = bi[0];
}

// x = mask*x + (1-mask)*one_hot_minus(x, hard_onehot)  ==> row permutation
__global__ void k_update_x(float* __restrict__ X, const int* __restrict__ am, int parity) {
    const int m = blockIdx.x;
    const int d = m & (Dd - 1);
    if ((d & 1) == parity) return;       // masked row unchanged
    const int u = threadIdx.x;
    const int a = am[m];
    const float v = X[(size_t)m * Kk + ((u + a) & (Kk - 1))];
    __syncthreads();
    X[(size_t)m * Kk + u] = v;
}

__global__ void k_lse(const float* __restrict__ bl, float* __restrict__ lse) {
    __shared__ float red[128];
    const int d = blockIdx.x, k = threadIdx.x;
    const float v = bl[(size_t)d * Kk + k];
    red[k] = v; __syncthreads();
    for (int off = 64; off; off >>= 1) {
        if (k < off) red[k] = fmaxf(red[k], red[k + off]);
        __syncthreads();
    }
    const float mx = red[0];
    __syncthreads();
    red[k] = __expf(v - mx); __syncthreads();
    for (int off = 64; off; off >>= 1) {
        if (k < off) red[k] += red[k + off];
        __syncthreads();
    }
    if (k == 0) lse[d] = mx + __logf(red[0]);
}

__global__ __launch_bounds__(256)
void k_final(const float* __restrict__ X, const float* __restrict__ bl,
             const float* __restrict__ lse, float* __restrict__ out) {
    __shared__ float red[256];
    const int b = blockIdx.x;
    float s = 0.0f;
    for (int e = threadIdx.x; e < Dd * Kk; e += 256) {
        const int d = e >> 7;
        s += (bl[e] - lse[d]) * X[(size_t)b * Dd * Kk + e];
    }
    red[threadIdx.x] = s; __syncthreads();
    for (int off = 128; off; off >>= 1) {
        if (threadIdx.x < off) red[threadIdx.x] += red[threadIdx.x + off];
        __syncthreads();
    }
    if (threadIdx.x == 0) out[b] = red[0];
}

// ---------------- host orchestration ----------------
static inline void launch_gemm(const float* A, const bf16* W, float* C, const float* aux,
                               int M, int N, int K, int mode, hipStream_t s) {
    dim3 grid(M / GBM, N / GBN);
    switch (mode) {
        case 0: k_gemm<0><<<grid, 256, 0, s>>>(A, W, C, aux, M, N, K); break;
        case 1: k_gemm<1><<<grid, 256, 0, s>>>(A, W, C, aux, M, N, K); break;
        case 2: k_gemm<2><<<grid, 256, 0, s>>>(A, W, C, aux, M, N, K); break;
        case 3: k_gemm<3><<<grid, 256, 0, s>>>(A, W, C, aux, M, N, K); break;
        default: k_gemm<4><<<grid, 256, 0, s>>>(A, W, C, aux, M, N, K); break;
    }
}

extern "C" void kernel_launch(void* const* d_in, const int* in_sizes, int n_in,
                              void* d_out, int out_size, void* d_ws, size_t ws_size,
                              hipStream_t stream) {
    (void)in_sizes; (void)n_in; (void)out_size; (void)ws_size;
    const int*   y       = (const int*)  d_in[0];
    const float* base_l  = (const float*)d_in[1];   // (D,K)
    const float* embed_W = (const float*)d_in[2];   // (NF, 512, 128)
    const float* pos_e   = (const float*)d_in[3];   // (NF, 256, 512)
    const float* qkv_W   = (const float*)d_in[4];   // (NF, 4, 1536, 512)
    const float* out_W   = (const float*)d_in[5];   // (NF, 4, 512, 512)
    const float* up_W    = (const float*)d_in[6];   // (NF, 4, 2048, 512)
    const float* down_W  = (const float*)d_in[7];   // (NF, 4, 512, 2048)
    const float* sk_a    = (const float*)d_in[8];   // (NF, 2)
    const float* head_W  = (const float*)d_in[9];   // (NF, 128, 512)
    const float* head_b  = (const float*)d_in[10];  // (NF, 128)
    float* out = (float*)d_out;

    char* ws = (char*)d_ws;
    float* X    = (float*)(ws + 0);                       // 8192x128  (4MB)
    float* H    = (float*)(ws + (size_t)4  * 1048576);    // 8192x512  (16MB)
    float* S0   = (float*)(ws + (size_t)20 * 1048576);    // 16MB
    float* S1   = (float*)(ws + (size_t)36 * 1048576);    // 16MB
    float* ACTA = (float*)(ws + (size_t)52 * 1048576);    // 8192x512  (16MB)
    float* BIG  = (float*)(ws + (size_t)68 * 1048576);    // 8192x2048 (64MB)
    int*   AMX  = (int*)  (ws + (size_t)132 * 1048576);   // 8192 ints
    float* LSE  = (float*)(ws + (size_t)132 * 1048576 + 65536); // 256 floats
    bf16*  WB   = (bf16*) (ws + (size_t)136 * 1048576);   // bf16 weights (~102MB)

    // bf16 weight pool element offsets
    const size_t o_embed = 0;
    const size_t o_qkv   = o_embed + (size_t)NFc * DIMc * Kk;
    const size_t o_out   = o_qkv   + (size_t)NFc * DEPTHc * 1536 * 512;
    const size_t o_up    = o_out   + (size_t)NFc * DEPTHc * 512 * 512;
    const size_t o_dn    = o_up    + (size_t)NFc * DEPTHc * 2048 * 512;
    const size_t o_hd    = o_dn    + (size_t)NFc * DEPTHc * 512 * 2048;

    // ---- pre-convert all weights f32 -> bf16 (element counts are /1024) ----
    auto cvt = [&](const float* src, bf16* dst, size_t n) {
        k_cvt_bf16<<<(unsigned)(n / 1024), 256, 0, stream>>>(src, dst);
    };
    cvt(embed_W, WB + o_embed, (size_t)NFc * DIMc * Kk);
    cvt(qkv_W,   WB + o_qkv,   (size_t)NFc * DEPTHc * 1536 * 512);
    cvt(out_W,   WB + o_out,   (size_t)NFc * DEPTHc * 512 * 512);
    cvt(up_W,    WB + o_up,    (size_t)NFc * DEPTHc * 2048 * 512);
    cvt(down_W,  WB + o_dn,    (size_t)NFc * DEPTHc * 512 * 2048);
    cvt(head_W,  WB + o_hd,    (size_t)NFc * Kk * DIMc);

    const size_t Hbytes = (size_t)Mrows * DIMc * sizeof(float);

    k_onehot<<<Mrows, Kk, 0, stream>>>(y, X);

    for (int i = NFc - 1; i >= 0; --i) {
        const int parity = i & 1;
        // xm = mask * x -> ACTA (M x 128)
        k_mask<<<Mrows, Kk, 0, stream>>>(X, ACTA, parity);
        // h = xm @ embed_W[i]^T + pos_embed[i]
        launch_gemm(ACTA, WB + o_embed + (size_t)i * DIMc * Kk, H,
                    pos_e + (size_t)i * Dd * DIMc, Mrows, DIMc, Kk, 1, stream);

        for (int b = 0; b < DEPTHc; ++b) {
            if (b >= 2) { // h += sigmoid(alpha)*skip ; b==2 pops S1, b==3 pops S0
                const float* sk = (b == 2) ? S1 : S0;
                k_skip_add<<<(Mrows * DIMc) / 256, 256, 0, stream>>>(
                    H, sk, sk_a + (size_t)i * 2 + (b - 2));
            }
            const size_t wb = (size_t)(i * DEPTHc + b);
            // attention
            k_rms<<<Mrows, 256, 0, stream>>>(H, ACTA, DIMc);
            launch_gemm(ACTA, WB + o_qkv + wb * 1536 * 512, BIG, nullptr,
                        Mrows, 1536, DIMc, 0, stream);
            k_rope<<<Mrows * HEADSc * 2, 32, 0, stream>>>(BIG);
            k_attn<<<Bb * HEADSc, 256, Dd * HDc * sizeof(float), stream>>>(BIG, ACTA);
            launch_gemm(ACTA, WB + o_out + wb * 512 * 512, H, nullptr,
                        Mrows, DIMc, DIMc, 3, stream);   // h += attn_out @ oW^T
            // mlp
            k_rms<<<Mrows, 256, 0, stream>>>(H, ACTA, DIMc);
            launch_gemm(ACTA, WB + o_up + wb * 2048 * 512, BIG, nullptr,
                        Mrows, 2048, DIMc, 4, stream);   // relu(x)^2 fused
            launch_gemm(BIG, WB + o_dn + wb * 512 * 2048, H, nullptr,
                        Mrows, DIMc, 2048, 3, stream);   // h += hidden @ dW^T
            if (b < 2) {
                float* dst = (b == 0) ? S0 : S1;
                hipMemcpyAsync(dst, H, Hbytes, hipMemcpyDeviceToDevice, stream);
            }
        }
        // logits = rms(h) @ head_W^T + head_b
        k_rms<<<Mrows, 256, 0, stream>>>(H, ACTA, DIMc);
        launch_gemm(ACTA, WB + o_hd + (size_t)i * Kk * DIMc, BIG,
                    head_b + (size_t)i * Kk, Mrows, Kk, DIMc, 2, stream);
        // loc forward value == hard one-hot of argmax -> row permutation update
        k_argmax<<<Mrows, Kk, 0, stream>>>(BIG, AMX);
        k_update_x<<<Mrows, Kk, 0, stream>>>(X, AMX, parity);
    }

    k_lse<<<Dd, Kk, 0, stream>>>(base_l, LSE);
    k_final<<<Bb, 256, 0, stream>>>(X, base_l, LSE, out);
}
